// MHCLayer_55319178772485
// MI455X (gfx1250) — matrix operand order repaired and checked by reading, not compiled, not run on hardware
//
#include <hip/hip_runtime.h>

// MHC layer for MI455X (gfx1250, wave32, WMMA bf16 16x16x32, TDM async tiles).
// Pipeline: wtrans (W->bf16 transposed) -> routing (mix/sinkhorn/x_in)
//           -> GEMM1 (x_in@W1, gelu, bf16 H) -> GEMM2 (H@W2 + fused res/post epilogue).

typedef unsigned int   uint_t;
typedef unsigned short ushort_t;
typedef __attribute__((ext_vector_type(16))) __bf16 v16bf;
typedef __attribute__((ext_vector_type(8)))  float  v8f;
typedef __attribute__((ext_vector_type(4)))  unsigned int v4u;
typedef __attribute__((ext_vector_type(8)))  int v8i;
typedef __attribute__((ext_vector_type(4)))  int v4i;

#define T_TOK 16384
#define C_DIM 768
#define DFF_DIM 3072

#if __has_builtin(__builtin_amdgcn_tensor_load_to_lds) && \
    __has_builtin(__builtin_amdgcn_s_wait_tensorcnt)
#define USE_TDM 1
#else
#define USE_TDM 0
#endif

#if USE_TDM
#if defined(__clang_major__) && (__clang_major__ >= 23)
#define TDM_LOAD(g0, g1, g2, g3) \
  __builtin_amdgcn_tensor_load_to_lds(g0, g1, g2, g3, (v8i){0,0,0,0,0,0,0,0}, 0)
#else
#define TDM_LOAD(g0, g1, g2, g3) \
  __builtin_amdgcn_tensor_load_to_lds(g0, g1, g2, g3, 0)
#endif
#endif

__device__ __forceinline__ ushort_t f2bf(float f) {
  uint_t u = __float_as_uint(f);
  u += 0x7FFFu + ((u >> 16) & 1u);   // round-to-nearest-even
  return (ushort_t)(u >> 16);
}

struct U8 { uint4 lo, hi; };

// A fragment: 16x32 bf16. lane 0-15: row=lane, K={kb..kb+7, kb+16..kb+23}, kb=0;
// lane 16-31: same rows, kb=8.  LDS row stride = 40 halves (80B, 16B-aligned).
__device__ __forceinline__ v16bf load_frag_a(const ushort_t* As, int lane, int rowBase) {
  int r  = rowBase + (lane & 15);
  int kb = (lane >> 4) << 3;               // 0 or 8
  const ushort_t* p = As + r * 40 + kb;
  U8 t;
  t.lo = *(const uint4*)(p);               // K = kb .. kb+7
  t.hi = *(const uint4*)(p + 16);          // K = kb+16 .. kb+23
  return __builtin_bit_cast(v16bf, t);
}

// B fragment: 32x16 bf16, stored transposed in LDS as Bs[n][k] (stride 40).
// lane 0-15: col=lane, K={0..7,16..23} ; lane 16-31: col=lane-16, K={8..15,24..31}.
__device__ __forceinline__ v16bf load_frag_b(const ushort_t* Bs, int lane, int colBase) {
  int n  = colBase + (lane & 15);
  int kb = (lane >> 4) << 3;               // 0 or 8
  const ushort_t* p = Bs + n * 40 + kb;
  U8 t;
  t.lo = *(const uint4*)(p);
  t.hi = *(const uint4*)(p + 16);
  return __builtin_bit_cast(v16bf, t);
}

#if USE_TDM
// Issue a TDM load of a 2D bf16 tile: 32 (k, contiguous) x 128 (rows, stride=K
// elements) into LDS at byte offset lds_off, hardware-padded to 40-half rows
// (pad_interval = 16 dwords, pad_amount = 4 dwords).
__device__ __forceinline__ void tdm_load_tile(unsigned lds_off,
                                              const ushort_t* gptr,
                                              int Kelems, int rowsTotal) {
  unsigned long long ga = (unsigned long long)(size_t)gptr;
  v4u g0;
  g0[0] = 1u;                                            // count=1 (valid), user mode
  g0[1] = lds_off;                                       // lds_addr (bytes)
  g0[2] = (unsigned)(ga & 0xFFFFFFFFu);                  // global_addr[31:0]
  g0[3] = (unsigned)((ga >> 32) & 0x01FFFFFFu) | (2u << 30); // addr[56:32] | type=2
  unsigned td0 = (unsigned)Kelems, td1 = (unsigned)rowsTotal;
  unsigned long long s0 = (unsigned long long)Kelems;    // dim0 stride (elems)
  v8i g1;
  g1[0] = (int)((1u << 16)        // data_size = 1 -> 2 bytes
              | (1u << 20)        // pad_enable
              | (3u << 22)        // pad_interval: 16 dwords (one 64B tile row)
              | (3u << 25));      // pad_amount: 4 dwords (16B)
  g1[1] = (int)((td0 & 0xFFFFu) << 16);                        // tensor_dim0.lo
  g1[2] = (int)((td0 >> 16) | ((td1 & 0xFFFFu) << 16));        // dim0.hi | dim1.lo
  g1[3] = (int)((td1 >> 16) | (32u << 16));                    // dim1.hi | tile_dim0=32
  g1[4] = (int)(128u);                                         // tile_dim1=128, tile_dim2=0
  g1[5] = (int)(unsigned)(s0 & 0xFFFFFFFFu);                   // dim0_stride[31:0]
  g1[6] = (int)(unsigned)((s0 >> 32) & 0xFFFFu);               // dim0_stride[47:32]
  g1[7] = 0;
  v4i gz = {0, 0, 0, 0};
  TDM_LOAD(g0, g1, gz, gz);
}
#endif

// Shared WMMA GEMM inner loop.  lds must hold 2 x (A 128x40 + B 128x40) halves
// (40960 bytes).  acc[4][2] covers the wave's 64x32 sub-tile of the 128x128 block.
__device__ __forceinline__ void gemm_core(const ushort_t* __restrict__ Ag,
                                          const ushort_t* __restrict__ Btg,
                                          int K, int Mtot, int Ntot,
                                          int m0, int n0, int tid,
                                          ushort_t* lds, v8f acc[4][2]) {
  const int lane = tid & 31;
  const int wave = tid >> 5;
  const int wm   = wave & 1;
  const int wn   = wave >> 1;
  const int nt   = K >> 5;

#if USE_TDM
  const unsigned lds0 = (unsigned)((size_t)(void*)lds & 0xFFFFFFFFu);
  if (wave == 0) {
    tdm_load_tile(lds0,         Ag  + (size_t)m0 * K, K, Mtot);
    tdm_load_tile(lds0 + 10240, Btg + (size_t)n0 * K, K, Ntot);
  }
  for (int i = 0; i < nt; ++i) {
    const int p = i & 1;
    if (wave == 0) {
      if (i + 1 < nt) {
        const unsigned boff = (unsigned)(p ^ 1) * 20480u;
        const int kk2 = (i + 1) << 5;
        tdm_load_tile(lds0 + boff,         Ag  + (size_t)m0 * K + kk2, K, Mtot);
        tdm_load_tile(lds0 + boff + 10240, Btg + (size_t)n0 * K + kk2, K, Ntot);
        __builtin_amdgcn_s_wait_tensorcnt((short)2);  // tile i fully in LDS
      } else {
        __builtin_amdgcn_s_wait_tensorcnt((short)0);
      }
    }
    __syncthreads();
    const ushort_t* As = lds + p * 10240;
    const ushort_t* Bs = As + 5120;
    v16bf af[4], bf2[2];
    #pragma unroll
    for (int a = 0; a < 4; ++a) af[a]  = load_frag_a(As, lane, wm * 64 + a * 16);
    #pragma unroll
    for (int b = 0; b < 2; ++b) bf2[b] = load_frag_b(Bs, lane, wn * 32 + b * 16);
    #pragma unroll
    for (int a = 0; a < 4; ++a)
      #pragma unroll
      for (int b = 0; b < 2; ++b)
        acc[a][b] = __builtin_amdgcn_wmma_f32_16x16x32_bf16(
            false, af[a], false, bf2[b], (short)0, acc[a][b], false, false);
    __syncthreads();
  }
#else
  // Fallback: synchronous single-buffer tiles via global->VGPR->LDS.
  ushort_t* As = lds;
  ushort_t* Bs = lds + 5120;
  for (int kk = 0; kk < K; kk += 32) {
    __syncthreads();
    #pragma unroll
    for (int i = 0; i < 2; ++i) {
      int idx = tid + i * 256;
      int r = idx >> 2, q = idx & 3;
      *(uint4*)(As + r * 40 + q * 8) = *(const uint4*)(Ag  + (size_t)(m0 + r) * K + kk + q * 8);
      *(uint4*)(Bs + r * 40 + q * 8) = *(const uint4*)(Btg + (size_t)(n0 + r) * K + kk + q * 8);
    }
    __syncthreads();
    v16bf af[4], bf2[2];
    #pragma unroll
    for (int a = 0; a < 4; ++a) af[a]  = load_frag_a(As, lane, wm * 64 + a * 16);
    #pragma unroll
    for (int b = 0; b < 2; ++b) bf2[b] = load_frag_b(Bs, lane, wn * 32 + b * 16);
    #pragma unroll
    for (int a = 0; a < 4; ++a)
      #pragma unroll
      for (int b = 0; b < 2; ++b)
        acc[a][b] = __builtin_amdgcn_wmma_f32_16x16x32_bf16(
            false, af[a], false, bf2[b], (short)0, acc[a][b], false, false);
  }
#endif
}

// ---------------------------------------------------------------------------
// Kernel 0: transpose + convert weights to bf16:  Wt[n][k] = bf16(W[k][n])
// ---------------------------------------------------------------------------
__global__ __launch_bounds__(256) void wtrans_kernel(const float* __restrict__ W,
                                                     ushort_t* __restrict__ Wt,
                                                     int R, int Ccols) {
  int idx = blockIdx.x * 256 + threadIdx.x;
  if (idx >= R * Ccols) return;
  int k = idx / Ccols;
  int n = idx - k * Ccols;
  Wt[(size_t)n * R + k] = f2bf(W[idx]);
}

// ---------------------------------------------------------------------------
// Kernel 1: routing. 64 tokens per block, 256 threads.
// ---------------------------------------------------------------------------
__global__ __launch_bounds__(256) void routing_kernel(
    const float* __restrict__ xg, const float* __restrict__ phig,
    const float* __restrict__ bg, const float* __restrict__ a_pre,
    const float* __restrict__ a_post, const float* __restrict__ a_res,
    ushort_t* __restrict__ xing, float* __restrict__ hresg,
    float* __restrict__ hpostg)
{
  __shared__ float xs[64 * 132];
  __shared__ float ps[3072];
  __shared__ float ssq[256];
  __shared__ float invr[64];
  __shared__ float mixs[64 * 24];
  __shared__ float hpre[64 * 4];

  const int tid  = threadIdx.x;
  const int tok0 = blockIdx.x * 64;
  const int t    = tid >> 2;
  const int seg  = tid & 3;
  const int j0   = seg * 6;

  float acc[6] = {0.f, 0.f, 0.f, 0.f, 0.f, 0.f};
  float sq = 0.f;

  for (int kc = 0; kc < 24; ++kc) {
    __syncthreads();
    {
      const float4* src = (const float4*)(phig + (size_t)kc * 128 * 24);
      float4* dst = (float4*)ps;
      #pragma unroll
      for (int s = 0; s < 3; ++s) dst[tid + s * 256] = src[tid + s * 256];
    }
    {
      const float* src = xg + (size_t)(tok0 + t) * 3072 + kc * 128 + seg * 32;
      float* dst = xs + t * 132 + seg * 32;
      #pragma unroll
      for (int u = 0; u < 8; ++u) {
        float4 v = *(const float4*)(src + u * 4);
        *(float4*)(dst + u * 4) = v;
        sq += v.x * v.x + v.y * v.y + v.z * v.z + v.w * v.w;
      }
    }
    __syncthreads();
    for (int k = 0; k < 128; ++k) {
      float xv = xs[t * 132 + k];
      const float* pr = ps + k * 24 + j0;
      #pragma unroll
      for (int p = 0; p < 6; ++p) acc[p] += xv * pr[p];
    }
  }

  ssq[tid] = sq;
  __syncthreads();
  if (tid < 64) {
    float s = ssq[tid * 4] + ssq[tid * 4 + 1] + ssq[tid * 4 + 2] + ssq[tid * 4 + 3];
    invr[tid] = rsqrtf(s * (1.f / 3072.f) + 1e-6f);
  }
  __syncthreads();
  {
    float iv = invr[t];
    #pragma unroll
    for (int p = 0; p < 6; ++p) mixs[t * 24 + j0 + p] = acc[p] * iv;
  }
  __syncthreads();

  if (tid < 64) {
    int tok = tok0 + tid;
    float ap = a_pre[0], apo = a_post[0], ar = a_res[0];
    float hp[4], hq[4], r[16];
    #pragma unroll
    for (int i = 0; i < 4; ++i) {
      hp[i] = 1.f / (1.f + __expf(-(mixs[tid * 24 + i] * ap + bg[i])));
      hq[i] = 2.f / (1.f + __expf(-(mixs[tid * 24 + 4 + i] * apo + bg[4 + i])));
    }
    #pragma unroll
    for (int e = 0; e < 16; ++e) r[e] = mixs[tid * 24 + 8 + e] * ar + bg[8 + e];
    #pragma unroll
    for (int i = 0; i < 4; ++i) {
      float mx = fmaxf(fmaxf(r[i*4], r[i*4+1]), fmaxf(r[i*4+2], r[i*4+3]));
      #pragma unroll
      for (int j = 0; j < 4; ++j) r[i*4+j] = __expf(r[i*4+j] - mx);
    }
    for (int it = 0; it < 20; ++it) {
      #pragma unroll
      for (int i = 0; i < 4; ++i) {
        float is = 1.f / (r[i*4] + r[i*4+1] + r[i*4+2] + r[i*4+3] + 1e-6f);
        #pragma unroll
        for (int j = 0; j < 4; ++j) r[i*4+j] *= is;
      }
      #pragma unroll
      for (int j = 0; j < 4; ++j) {
        float is = 1.f / (r[j] + r[4+j] + r[8+j] + r[12+j] + 1e-6f);
        #pragma unroll
        for (int i = 0; i < 4; ++i) r[i*4+j] *= is;
      }
    }
    #pragma unroll
    for (int i = 0; i < 4; ++i) {
      hpre[tid * 4 + i] = hp[i];
      hpostg[(size_t)tok * 4 + i] = hq[i];
    }
    #pragma unroll
    for (int e = 0; e < 16; ++e) hresg[(size_t)tok * 16 + e] = r[e];
  }
  __syncthreads();

  {
    float h0 = hpre[t*4+0], h1 = hpre[t*4+1], h2 = hpre[t*4+2], h3 = hpre[t*4+3];
    const float* xr = xg + (size_t)(tok0 + t) * 3072;
    ushort_t* xo = xing + (size_t)(tok0 + t) * 768 + seg * 192;
    int c0 = seg * 192;
    for (int u = 0; u < 48; ++u) {
      int c = c0 + u * 4;
      float4 a0 = *(const float4*)(xr + c);
      float4 a1 = *(const float4*)(xr + 768 + c);
      float4 a2 = *(const float4*)(xr + 1536 + c);
      float4 a3 = *(const float4*)(xr + 2304 + c);
      ushort4 pk;
      pk.x = f2bf(h0 * a0.x + h1 * a1.x + h2 * a2.x + h3 * a3.x);
      pk.y = f2bf(h0 * a0.y + h1 * a1.y + h2 * a2.y + h3 * a3.y);
      pk.z = f2bf(h0 * a0.z + h1 * a1.z + h2 * a2.z + h3 * a3.z);
      pk.w = f2bf(h0 * a0.w + h1 * a1.w + h2 * a2.w + h3 * a3.w);
      *(ushort4*)(xo + u * 4) = pk;
    }
  }
}

// ---------------------------------------------------------------------------
// Kernel 2: GEMM1  H = gelu(x_in @ W1 + b1), bf16 out.  M=16384,K=768,N=3072.
// ---------------------------------------------------------------------------
__global__ __launch_bounds__(256) void gemm1_kernel(
    const ushort_t* __restrict__ Ag, const ushort_t* __restrict__ Btg,
    const float* __restrict__ b1g, ushort_t* __restrict__ Hg)
{
  const int K = 768, Ncols = 3072;
  __shared__ __align__(16) ushort_t lds[2 * 10240];   // 40KB: 2x(A+B) tiles

  const int tid  = threadIdx.x;
  const int lane = tid & 31;
  const int wave = tid >> 5;
  const int wm   = wave & 1;
  const int wn   = wave >> 1;
  const int m0   = (blockIdx.x & 127) * 128;
  const int n0   = (blockIdx.x >> 7) * 128;

  v8f acc[4][2];
  #pragma unroll
  for (int a = 0; a < 4; ++a)
    #pragma unroll
    for (int b = 0; b < 2; ++b) acc[a][b] = (v8f){0,0,0,0,0,0,0,0};

  gemm_core(Ag, Btg, K, T_TOK, Ncols, m0, n0, tid, lds, acc);

  // epilogue: bias + exact gelu, stage bf16 tile in LDS, coalesced store
  __syncthreads();
  ushort_t* Cst = lds;                       // 128 x 128, stride 136 (34816B)
  #pragma unroll
  for (int a = 0; a < 4; ++a)
    #pragma unroll
    for (int b = 0; b < 2; ++b) {
      int c = wn * 32 + b * 16 + (lane & 15);
      float bias = b1g[n0 + c];
      #pragma unroll
      for (int v = 0; v < 8; ++v) {
        int r = wm * 64 + a * 16 + ((lane >> 4) << 3) + v;
        float h = acc[a][b][v] + bias;
        float g = 0.5f * h * (1.f + erff(h * 0.70710678118f));
        Cst[r * 136 + c] = f2bf(g);
      }
    }
  __syncthreads();
  {
    int r = tid >> 1, half = tid & 1;
    const ushort_t* src = Cst + r * 136 + half * 64;
    ushort_t* dst = Hg + (size_t)(m0 + r) * Ncols + n0 + half * 64;
    #pragma unroll
    for (int q = 0; q < 8; ++q)
      *(uint4*)(dst + q * 8) = *(const uint4*)(src + q * 8);
  }
}

// ---------------------------------------------------------------------------
// Kernel 3: GEMM2  f = H @ W2 + b2; out = h_res@x + h_post*f.  M=16384,K=3072,N=768.
// ---------------------------------------------------------------------------
__global__ __launch_bounds__(256) void gemm2_kernel(
    const ushort_t* __restrict__ Ag, const ushort_t* __restrict__ Btg,
    const float* __restrict__ b2g, const float* __restrict__ xg,
    const float* __restrict__ hresg, const float* __restrict__ hpostg,
    float* __restrict__ outg)
{
  const int K = 3072;
  __shared__ __align__(16) ushort_t lds[2 * 10240];

  const int tid  = threadIdx.x;
  const int lane = tid & 31;
  const int wave = tid >> 5;
  const int wm   = wave & 1;
  const int wn   = wave >> 1;
  const int m0   = (blockIdx.x & 127) * 128;
  const int n0   = (blockIdx.x >> 7) * 128;

  v8f acc[4][2];
  #pragma unroll
  for (int a = 0; a < 4; ++a)
    #pragma unroll
    for (int b = 0; b < 2; ++b) acc[a][b] = (v8f){0,0,0,0,0,0,0,0};

  gemm_core(Ag, Btg, K, T_TOK, 768, m0, n0, tid, lds, acc);

  // stage h_res (16) + h_post (4) per row into LDS
  __syncthreads();
  float* hsf = (float*)lds;                  // 128 x 20 f32 (10240B)
  for (int i = tid; i < 128 * 20; i += 256) {
    int r = i / 20, q = i - r * 20;
    hsf[i] = (q < 16) ? hresg[(size_t)(m0 + r) * 16 + q]
                      : hpostg[(size_t)(m0 + r) * 4 + (q - 16)];
  }
  __syncthreads();

  #pragma unroll
  for (int a = 0; a < 4; ++a)
    #pragma unroll
    for (int b = 0; b < 2; ++b) {
      int c = n0 + wn * 32 + b * 16 + (lane & 15);
      float bias = b2g[c];
      #pragma unroll
      for (int v = 0; v < 8; ++v) {
        int lr = wm * 64 + a * 16 + ((lane >> 4) << 3) + v;
        int m  = m0 + lr;
        float f = acc[a][b][v] + bias;
        const float* xr = xg + (size_t)m * 3072 + c;
        float x0 = xr[0], x1 = xr[768], x2 = xr[1536], x3 = xr[2304];
        const float* h = &hsf[lr * 20];
        float* o = outg + (size_t)m * 3072 + c;
        #pragma unroll
        for (int i = 0; i < 4; ++i) {
          float rg = h[i*4+0]*x0 + h[i*4+1]*x1 + h[i*4+2]*x2 + h[i*4+3]*x3;
          o[i * 768] = rg + h[16 + i] * f;
        }
      }
    }
}

// ---------------------------------------------------------------------------
extern "C" void kernel_launch(void* const* d_in, const int* in_sizes, int n_in,
                              void* d_out, int out_size, void* d_ws, size_t ws_size,
                              hipStream_t stream) {
  (void)in_sizes; (void)n_in; (void)out_size; (void)ws_size;
  const float* xg    = (const float*)d_in[0];
  const float* phig  = (const float*)d_in[1];
  const float* bg    = (const float*)d_in[2];
  const float* apre  = (const float*)d_in[3];
  const float* apost = (const float*)d_in[4];
  const float* ares  = (const float*)d_in[5];
  const float* W1    = (const float*)d_in[6];
  const float* b1    = (const float*)d_in[7];
  const float* W2    = (const float*)d_in[8];
  const float* b2    = (const float*)d_in[9];
  float* outg = (float*)d_out;

  char* ws = (char*)d_ws;
  size_t off = 0;
  ushort_t* xin  = (ushort_t*)(ws + off); off += (size_t)T_TOK * C_DIM * 2;
  ushort_t* Hbuf = (ushort_t*)(ws + off); off += (size_t)T_TOK * DFF_DIM * 2;
  ushort_t* Wt1  = (ushort_t*)(ws + off); off += (size_t)DFF_DIM * C_DIM * 2;
  ushort_t* Wt2  = (ushort_t*)(ws + off); off += (size_t)C_DIM * DFF_DIM * 2;
  float*    hres = (float*)(ws + off);    off += (size_t)T_TOK * 16 * 4;
  float*    hpost= (float*)(ws + off);    off += (size_t)T_TOK * 4 * 4;

  int wn = C_DIM * DFF_DIM;
  wtrans_kernel<<<(wn + 255) / 256, 256, 0, stream>>>(W1, Wt1, C_DIM, DFF_DIM);
  wtrans_kernel<<<(wn + 255) / 256, 256, 0, stream>>>(W2, Wt2, DFF_DIM, C_DIM);

  routing_kernel<<<T_TOK / 64, 256, 0, stream>>>(xg, phig, bg, apre, apost, ares,
                                                 xin, hres, hpost);

  gemm1_kernel<<<(T_TOK / 128) * (DFF_DIM / 128), 256, 0, stream>>>(xin, Wt1, b1, Hbuf);
  gemm2_kernel<<<(T_TOK / 128) * (C_DIM / 128), 256, 0, stream>>>(Hbuf, Wt2, b2,
                                                                  xg, hres, hpost, outg);
}
